// Net_20779051778391
// MI455X (gfx1250) — matrix operand order repaired and checked by reading, not compiled, not run on hardware
//
#include <hip/hip_runtime.h>
#include <hip/hip_bf16.h>
#include <stdint.h>
#include <stddef.h>

#define Hh 64
#define Nn 8192
#define Ee 131072
#define Tt 12

typedef __attribute__((ext_vector_type(16))) __bf16 v16bf;
typedef __attribute__((ext_vector_type(8)))  __bf16 v8bf;
typedef __attribute__((ext_vector_type(8)))  float  v8f;

static __device__ __forceinline__ v16bf cat8(v8bf a, v8bf b) {
  v16bf r;
#pragma unroll
  for (int i = 0; i < 8; i++) { r[i] = a[i]; r[8 + i] = b[i]; }
  return r;
}
static __device__ __forceinline__ v8f vzero8() {
  v8f r;
#pragma unroll
  for (int i = 0; i < 8; i++) r[i] = 0.f;
  return r;
}
static __device__ __forceinline__ void atomAddF(float* p, float v) {
  __hip_atomic_fetch_add(p, v, __ATOMIC_RELAXED, __HIP_MEMORY_SCOPE_AGENT);
}

// ---------------------------------------------------------------- fp32 -> bf16
__global__ void k_f2bf(const float* __restrict__ src, __bf16* __restrict__ dst, size_t n) {
  size_t stride = (size_t)gridDim.x * blockDim.x;
  for (size_t i = (size_t)blockIdx.x * blockDim.x + threadIdx.x; i < n; i += stride)
    dst[i] = (__bf16)src[i];
}

// ---------------------------------------------------------------- GRU + output head
// One thread per node; h kept in LDS (per-thread private row); weights via uniform
// (scalar-cache) loads. Writes his -> out[0..N-1] (pred_result, skip=True path).
__global__ __launch_bounds__(64) void k_gru(
    const float* __restrict__ node_attrs,  // [T,N]
    const float* __restrict__ w_ih,        // [192] ([3H,1])
    const float* __restrict__ w_hh,        // [192,64]
    const float* __restrict__ b_ih, const float* __restrict__ b_hh,
    const float* __restrict__ w_lout, const float* __restrict__ b_lout,
    float* __restrict__ his) {
  __shared__ float hbuf[2][64][65];
  int tid = threadIdx.x;
  int n = blockIdx.x * 64 + tid;
  for (int k = 0; k < 64; k++) hbuf[0][tid][k] = 0.f;
  for (int t = 0; t < Tt; t++) {
    float x = node_attrs[(size_t)t * Nn + n];
    const float* hrow = hbuf[t & 1][tid];
    float* hnew = hbuf[(t + 1) & 1][tid];
    for (int j = 0; j < 64; j++) {
      float hr = b_hh[j], hz = b_hh[64 + j], hn = b_hh[128 + j];
      for (int k = 0; k < 64; k++) {
        float hk = hrow[k];
        hr += w_hh[j * 64 + k] * hk;
        hz += w_hh[(64 + j) * 64 + k] * hk;
        hn += w_hh[(128 + j) * 64 + k] * hk;
      }
      float xr = b_ih[j] + x * w_ih[j];
      float xz = b_ih[64 + j] + x * w_ih[64 + j];
      float xn = b_ih[128 + j] + x * w_ih[128 + j];
      float r = 1.f / (1.f + __expf(-(xr + hr)));
      float z = 1.f / (1.f + __expf(-(xz + hz)));
      float nn2 = tanhf(xn + r * hn);
      hnew[j] = (1.f - z) * nn2 + z * hrow[j];
    }
  }
  const float* hf = hbuf[0][tid];  // T=12 even -> final state in buffer 0
  float acc = b_lout[0];
  for (int k = 0; k < 64; k++) acc += w_lout[k] * hf[k];
  his[n] = acc;
}

// ---------------------------------------------------------------- edge block
// 4 waves/block, 16 edges/wave. e_in (7*64=448 bf16) staged K-contiguous in LDS;
// new_e = relu(W_eb @ e_in + b) via v_wmma_f32_16x16x32_bf16 (4 out-tiles x 14 K-chunks).
// Scatter-add into sent/recvd with global_atomic_add_f32.
__global__ __launch_bounds__(128) void k_edge(
    const float* __restrict__ ea_t,   // edge_attrs[t] : [E]
    const float* __restrict__ na_t,   // node_attrs[t] : [N]
    const int* __restrict__ sendi, const int* __restrict__ recvi,
    const __bf16* __restrict__ he_in,  // [E,64]
    const float* __restrict__ hx,      // [N,64]
    const float* __restrict__ gl,      // [64]
    const float* __restrict__ w_ee, const float* __restrict__ b_ee,
    const float* __restrict__ w_ne, const float* __restrict__ b_ne,
    const __bf16* __restrict__ w_eb_bf,  // [64,448]
    const float* __restrict__ b_eb,
    __bf16* __restrict__ new_e,          // [E,64]
    float* __restrict__ sent_acc, float* __restrict__ recvd_acc) {
  __shared__ alignas(32) __bf16 lds[4][16][448];
  int wid = threadIdx.x >> 5, lane = threadIdx.x & 31;
  int el = lane & 15, hi = lane >> 4;
  int e0 = (blockIdx.x * 4 + wid) * 16;
  int e = e0 + el;
  float ea = ea_t[e];
  int s = sendi[e], r = recvi[e];
  float na_s = na_t[s], na_r = na_t[r];
  __bf16* row = &lds[wid][el][0];
  for (int j = 0; j < 32; j++) {
    int h = hi * 32 + j;
    row[0 * 64 + h] = (__bf16)fmaxf(ea * w_ee[h] + b_ee[h], 0.f);
    row[1 * 64 + h] = (__bf16)fmaxf(na_s * w_ne[h] + b_ne[h], 0.f);
    row[2 * 64 + h] = (__bf16)fmaxf(na_r * w_ne[h] + b_ne[h], 0.f);
    row[3 * 64 + h] = he_in[(size_t)e * 64 + h];
    row[4 * 64 + h] = (__bf16)hx[(size_t)s * 64 + h];
    row[5 * 64 + h] = (__bf16)hx[(size_t)r * 64 + h];
    row[6 * 64 + h] = (__bf16)gl[h];
  }
  __syncthreads();

  v8f acc[4];
#pragma unroll
  for (int ot = 0; ot < 4; ot++) acc[ot] = vzero8();
  for (int kc = 0; kc < 14; kc++) {
    int k0 = kc * 32;
    v16bf bm = *(const v16bf*)&lds[wid][el][k0 + hi * 16];
#pragma unroll
    for (int ot = 0; ot < 4; ot++) {
      const __bf16* ar = w_eb_bf + (size_t)(ot * 16 + el) * 448 + k0 + hi * 16;
      v8bf a0 = *(const v8bf*)(ar);
      v8bf a1 = *(const v8bf*)(ar + 8);
      acc[ot] = __builtin_amdgcn_wmma_f32_16x16x32_bf16(
          false, cat8(a0, a1), false, bm, (short)0, acc[ot], false, false);
    }
  }
#pragma unroll
  for (int ot = 0; ot < 4; ot++) {
    int ob = ot * 16 + hi * 8;
    v8bf pk;
#pragma unroll
    for (int g = 0; g < 8; g++) {
      float v = fmaxf(acc[ot][g] + b_eb[ob + g], 0.f);
      pk[g] = (__bf16)v;
      atomAddF(&sent_acc[(size_t)s * 64 + ob + g], v);
      atomAddF(&recvd_acc[(size_t)r * 64 + ob + g], v);
    }
    *(v8bf*)&new_e[(size_t)e * 64 + ob] = pk;
  }
}

// ---------------------------------------------------------------- node block
// n_in = [nx, hx, recvd, sent, gl] (320) -> new_n = relu(W_nb @ n_in + b); td = new_n - hx.
__global__ __launch_bounds__(128) void k_node(
    const float* __restrict__ na_t,  // node_attrs[t] : [N]
    const float* __restrict__ hx, const float* __restrict__ recvd,
    const float* __restrict__ sentv, const float* __restrict__ gl,
    const float* __restrict__ w_ne, const float* __restrict__ b_ne,
    const __bf16* __restrict__ w_nb_bf,  // [64,320]
    const float* __restrict__ b_nb,
    float* __restrict__ new_n, float* __restrict__ tds) {
  __shared__ alignas(32) __bf16 lds[4][16][320];
  int wid = threadIdx.x >> 5, lane = threadIdx.x & 31;
  int el = lane & 15, hi = lane >> 4;
  int n0 = (blockIdx.x * 4 + wid) * 16;
  int n = n0 + el;
  float xna = na_t[n];
  __bf16* row = &lds[wid][el][0];
  for (int j = 0; j < 32; j++) {
    int h = hi * 32 + j;
    row[h] = (__bf16)fmaxf(xna * w_ne[h] + b_ne[h], 0.f);
    row[64 + h] = (__bf16)hx[(size_t)n * 64 + h];
    row[128 + h] = (__bf16)recvd[(size_t)n * 64 + h];
    row[192 + h] = (__bf16)sentv[(size_t)n * 64 + h];
    row[256 + h] = (__bf16)gl[h];
  }
  __syncthreads();

  v8f acc[4];
#pragma unroll
  for (int ot = 0; ot < 4; ot++) acc[ot] = vzero8();
  for (int kc = 0; kc < 10; kc++) {
    int k0 = kc * 32;
    v16bf bm = *(const v16bf*)&lds[wid][el][k0 + hi * 16];
#pragma unroll
    for (int ot = 0; ot < 4; ot++) {
      const __bf16* ar = w_nb_bf + (size_t)(ot * 16 + el) * 320 + k0 + hi * 16;
      v8bf a0 = *(const v8bf*)(ar);
      v8bf a1 = *(const v8bf*)(ar + 8);
      acc[ot] = __builtin_amdgcn_wmma_f32_16x16x32_bf16(
          false, cat8(a0, a1), false, bm, (short)0, acc[ot], false, false);
    }
  }
#pragma unroll
  for (int ot = 0; ot < 4; ot++) {
    int ob = ot * 16 + hi * 8;
    v8f vout, tout;
#pragma unroll
    for (int g = 0; g < 8; g++) {
      float v = fmaxf(acc[ot][g] + b_nb[ob + g], 0.f);
      vout[g] = v;
      tout[g] = v - hx[(size_t)n * 64 + ob + g];
    }
    *(v8f*)&new_n[(size_t)n * 64 + ob] = vout;
    *(v8f*)&tds[(size_t)n * 64 + ob] = tout;
  }
}

// ---------------------------------------------------------------- column sums (for means)
__global__ void k_colsum_f32(const float* __restrict__ src, int rows, float* __restrict__ out64) {
  int c = threadIdx.x & 63;
  int rs = blockIdx.x * (blockDim.x >> 6) + (threadIdx.x >> 6);
  int rstride = gridDim.x * (blockDim.x >> 6);
  float acc = 0.f;
  for (int r = rs; r < rows; r += rstride) acc += src[(size_t)r * 64 + c];
  atomAddF(&out64[c], acc);
}
__global__ void k_colsum_bf16(const __bf16* __restrict__ src, int rows, float* __restrict__ out64) {
  int c = threadIdx.x & 63;
  int rs = blockIdx.x * (blockDim.x >> 6) + (threadIdx.x >> 6);
  int rstride = gridDim.x * (blockDim.x >> 6);
  float acc = 0.f;
  for (int r = rs; r < rows; r += rstride) acc += (float)src[(size_t)r * 64 + c];
  atomAddF(&out64[c], acc);
}

// ---------------------------------------------------------------- global block (tiny)
__global__ void k_global(const float* __restrict__ sums, float* __restrict__ gl,
                         const float* __restrict__ w_gb, const float* __restrict__ b_gb) {
  __shared__ float gin[192];
  int h = threadIdx.x;  // 64 threads
  gin[h] = sums[h] * (1.0f / (float)Nn);
  gin[64 + h] = sums[64 + h] * (1.0f / (float)Ee);
  gin[128 + h] = gl[h];
  __syncthreads();
  float acc = b_gb[h];
  for (int j = 0; j < 192; j++) acc += w_gb[h * 192 + j] * gin[j];
  gl[h] = fmaxf(acc, 0.f);
}

// ---------------------------------------------------------------- new_n -> new_nT (bf16)
__global__ void k_transpose(const float* __restrict__ nn, __bf16* __restrict__ nT) {
  int i = blockIdx.x * blockDim.x + threadIdx.x;  // over 64*N, contiguous in n
  int h = i >> 13;
  int n = i & (Nn - 1);
  nT[i] = (__bf16)nn[(size_t)n * 64 + h];
}

// ---------------------------------------------------------------- sd = coeff * sp_L @ new_n
// 8 waves/block; each wave owns one 16-row m-tile and ALL 64 output columns
// (4 independent WMMA accumulator chains). sp_L (bf16, L2-resident) is streamed
// exactly once per step; the [64x32] B slice of new_nT per K-chunk is staged in
// LDS (double-buffered, 80B row stride -> conflict-free ds_load_b128) and shared
// by all 8 waves. A-stream kept ahead of the WMMAs with global_prefetch_b8.
__global__ __launch_bounds__(256) void k_sd(
    const __bf16* __restrict__ spl,  // [N,N] bf16
    const __bf16* __restrict__ nT,   // [64,N] bf16
    const float* __restrict__ coeff,
    float* __restrict__ sds) {       // [N,64]
  __shared__ alignas(32) __bf16 bsl[2][64][40];  // 40-elem (80B) stride: bank-spread
  int tid = threadIdx.x;
  int wid = tid >> 5, lane = tid & 31;
  int el = lane & 15, hi = lane >> 4;
  int m0 = (blockIdx.x * 8 + wid) * 16;  // 64 blocks x 8 waves = 512 m-tiles
  int srow = tid >> 2;                   // staging: 4 threads x 16B per nT row
  int soff = (tid & 3) * 8;
  float c0 = coeff[0];

  v8f acc[4];
#pragma unroll
  for (int ot = 0; ot < 4; ot++) acc[ot] = vzero8();
  const __bf16* arow = spl + (size_t)(m0 + el) * Nn + hi * 8;
  const __bf16* brow = nT + (size_t)srow * Nn + soff;

  // stage K-chunk 0 into buffer 0
  *(v8bf*)&bsl[0][srow][soff] = *(const v8bf*)(brow);
  __syncthreads();

  for (int kc = 0; kc < Nn / 32; kc++) {
    int k0 = kc * 32;
    int cur = kc & 1;
    if (kc + 1 < Nn / 32)  // stage next chunk into the other buffer
      *(v8bf*)&bsl[cur ^ 1][srow][soff] = *(const v8bf*)(brow + k0 + 32);
    // prefetch the A stream 8 chunks (512B) ahead
    __builtin_prefetch((const void*)(arow + k0 + 8 * 32), 0, 0);
    v8bf a0 = *(const v8bf*)(arow + k0);
    v8bf a1 = *(const v8bf*)(arow + k0 + 16);
    v16bf am = cat8(a0, a1);
#pragma unroll
    for (int ot = 0; ot < 4; ot++) {
      v8bf b0 = *(const v8bf*)&bsl[cur][ot * 16 + el][hi * 8];
      v8bf b1 = *(const v8bf*)&bsl[cur][ot * 16 + el][hi * 8 + 16];
      acc[ot] = __builtin_amdgcn_wmma_f32_16x16x32_bf16(
          false, am, false, cat8(b0, b1), (short)0, acc[ot], false, false);
    }
    __syncthreads();
  }
#pragma unroll
  for (int ot = 0; ot < 4; ot++) {
#pragma unroll
    for (int g = 0; g < 8; g++) {
      int m = m0 + g + 8 * hi;
      sds[(size_t)m * 64 + ot * 16 + el] = c0 * acc[ot][g];
    }
  }
}

// ================================================================ host orchestration
extern "C" void kernel_launch(void* const* d_in, const int* in_sizes, int n_in,
                              void* d_out, int out_size, void* d_ws, size_t ws_size,
                              hipStream_t stream) {
  (void)in_sizes; (void)n_in; (void)out_size;
  const float* node_attrs = (const float*)d_in[0];
  const float* edge_attrs = (const float*)d_in[1];
  const float* global_attr = (const float*)d_in[2];
  const float* sp_L = (const float*)d_in[3];
  const float* coeff = (const float*)d_in[4];
  const float* w_ee = (const float*)d_in[5];
  const float* b_ee = (const float*)d_in[6];
  const float* w_ne = (const float*)d_in[7];
  const float* b_ne = (const float*)d_in[8];
  const float* w_eb = (const float*)d_in[9];
  const float* b_eb = (const float*)d_in[10];
  const float* w_nb = (const float*)d_in[11];
  const float* b_nb = (const float*)d_in[12];
  const float* w_gb = (const float*)d_in[13];
  const float* b_gb = (const float*)d_in[14];
  const float* w_ih = (const float*)d_in[15];
  const float* w_hh = (const float*)d_in[16];
  const float* b_ih = (const float*)d_in[17];
  const float* b_hh = (const float*)d_in[18];
  const float* w_lout = (const float*)d_in[19];
  const float* b_lout = (const float*)d_in[20];
  const int* eidx = (const int*)d_in[21];
  float* out = (float*)d_out;

  // --- workspace carve-up (bf16 sp_L is the big one: L2-resident across all 12 steps)
  char* p = (char*)d_ws;
  auto take = [&](size_t bytes) -> char* {
    char* r = p;
    p += (bytes + 255) & ~(size_t)255;
    return r;
  };
  __bf16* spL_bf = (__bf16*)take((size_t)Nn * Nn * 2);
  __bf16* heA = (__bf16*)take((size_t)Ee * Hh * 2);
  __bf16* heB = (__bf16*)take((size_t)Ee * Hh * 2);
  float* hxA = (float*)take((size_t)Nn * Hh * 4);
  float* hxB = (float*)take((size_t)Nn * Hh * 4);
  float* sentb = (float*)take((size_t)Nn * Hh * 4);
  float* recvb = (float*)take((size_t)Nn * Hh * 4);
  __bf16* nT = (__bf16*)take((size_t)Hh * Nn * 2);
  __bf16* webbf = (__bf16*)take((size_t)64 * 448 * 2);
  __bf16* wnbbf = (__bf16*)take((size_t)64 * 320 * 2);
  float* gl = (float*)take(256);
  float* sums = (float*)take(512);
  if ((size_t)(p - (char*)d_ws) > ws_size) return;  // insufficient scratch

  // --- one-time per call: init states, convert sp_L + block weights to bf16, run GRU head
  hipMemsetAsync(hxA, 0, (size_t)Nn * Hh * 4, stream);
  hipMemsetAsync(heA, 0, (size_t)Ee * Hh * 2, stream);
  hipMemcpyAsync(gl, global_attr, Hh * 4, hipMemcpyDeviceToDevice, stream);
  k_f2bf<<<4096, 256, 0, stream>>>(sp_L, spL_bf, (size_t)Nn * Nn);
  k_f2bf<<<64, 256, 0, stream>>>(w_eb, webbf, (size_t)64 * 448);
  k_f2bf<<<64, 256, 0, stream>>>(w_nb, wnbbf, (size_t)64 * 320);
  k_gru<<<Nn / 64, 64, 0, stream>>>(node_attrs, w_ih, w_hh, b_ih, b_hh, w_lout, b_lout, out);

  // --- GN scan over T steps
  for (int t = 0; t < Tt; t++) {
    const __bf16* he_in = (t & 1) ? heB : heA;
    __bf16* he_out = (t & 1) ? heA : heB;
    const float* hx_in = (t & 1) ? hxB : hxA;
    float* hx_out = (t & 1) ? hxA : hxB;
    float* tds_t = out + Nn + (size_t)t * Nn * Hh;
    float* sds_t = out + Nn + (size_t)Tt * Nn * Hh + (size_t)t * Nn * Hh;

    hipMemsetAsync(sentb, 0, (size_t)Nn * Hh * 4, stream);
    hipMemsetAsync(recvb, 0, (size_t)Nn * Hh * 4, stream);
    hipMemsetAsync(sums, 0, 128 * 4, stream);

    k_edge<<<Ee / 64, 128, 0, stream>>>(edge_attrs + (size_t)t * Ee, node_attrs + (size_t)t * Nn,
                                        eidx, eidx + Ee, he_in, hx_in, gl,
                                        w_ee, b_ee, w_ne, b_ne, webbf, b_eb,
                                        he_out, sentb, recvb);
    k_node<<<Nn / 64, 128, 0, stream>>>(node_attrs + (size_t)t * Nn, hx_in, recvb, sentb, gl,
                                        w_ne, b_ne, wnbbf, b_nb, hx_out, tds_t);
    k_colsum_f32<<<64, 256, 0, stream>>>(hx_out, Nn, sums);
    k_colsum_bf16<<<256, 256, 0, stream>>>(he_out, Ee, sums + 64);
    k_global<<<1, 64, 0, stream>>>(sums, gl, w_gb, b_gb);
    k_transpose<<<(Nn * Hh) / 256, 256, 0, stream>>>(hx_out, nT);
    k_sd<<<64, 256, 0, stream>>>(spL_bf, nT, coeff, sds_t);
  }
}
